// MixtralSparseMoeBlock_23261542875579
// MI455X (gfx1250) — matrix operand order repaired and checked by reading, not compiled, not run on hardware
//
#include <hip/hip_runtime.h>
#include <hip/hip_bf16.h>

// ---------------- problem constants (from reference) ----------------
#define NE     8          // experts
#define TOPK   2
#define D_DIM  2048       // hidden
#define M_DIM  7168       // mlp
#define T_TOK  4096       // B*S
#define TR     (T_TOK * TOPK)   // 8192 rows total, fixed

// ---------------- GEMM tiling ----------------
#define TM 64             // rows per workgroup tile
#define TN 128            // cols per workgroup tile
#define KC 32             // k-chunk == WMMA K for bf16
#define SA  40            // LDS stride of A tile rows (multiple of 8 -> 16B aligned)
#define SBT 40            // LDS stride of transposed-B rows (k-contiguous)
#define NTHREADS 256      // 8 waves (wave32): 4 wave-rows x 2 wave-cols

typedef __attribute__((ext_vector_type(16))) __bf16 v16bf;
typedef __attribute__((ext_vector_type(8)))  float  v8f;

union FragQ { uint4 q[2]; v16bf v; };
union AccU  { v8f   v; float  e[8]; };

// pack two floats to packed bf16 dword; halves written directly so ISel can
// emit a single v_cvt_pk_bf16_f32 (no u16 extract / shift / or fixups).
__device__ __forceinline__ unsigned pack_bf16x2(float a, float b) {
  union { __bf16 h[2]; unsigned u; } p;
  p.h[0] = (__bf16)a;
  p.h[1] = (__bf16)b;
  return p.u;
}

// A fragment: ISA 7.12.2 16-bit A 16x32: lane m=L%16,h=L/16,
// e[0..7] = A[m][8h .. 8h+7], e[8..15] = A[m][16+8h .. 16+8h+7]
// -> two contiguous 16B LDS loads from row-major As[m][k].
__device__ __forceinline__ v16bf load_a_frag(const __bf16* As_w, int lane) {
  const int m = lane & 15, h2 = lane >> 4;
  const __bf16* p = As_w + m * SA + (h2 << 3);
  FragQ f;
  f.q[0] = *(const uint4*)(p);
  f.q[1] = *(const uint4*)(p + 16);
  return f.v;
}

// B fragment: lane n=L%16,h=L/16, e[0..7] = B[8h .. 8h+7][n], e[8..15] = B[16+8h .. 16+8h+7][n]
// -> with B stored transposed (Bt[n][k], k contiguous): two 16B LDS loads.
__device__ __forceinline__ v16bf load_b_frag(const __bf16* Bt, int ncol, int lane) {
  const int n = lane & 15, h2 = lane >> 4;
  const __bf16* p = Bt + (size_t)(ncol + n) * SBT + (h2 << 3);
  FragQ f;
  f.q[0] = *(const uint4*)(p);
  f.q[1] = *(const uint4*)(p + 16);
  return f.v;
}

// ---------------- router: one wave per token ----------------
__global__ __launch_bounds__(256) void moe_router(
    const float* __restrict__ x, const float* __restrict__ gw,
    int* __restrict__ topk_idx, float* __restrict__ topk_w)
{
  const int wave = threadIdx.x >> 5, lane = threadIdx.x & 31;
  const int t = blockIdx.x * 8 + wave;
  float acc[NE];
#pragma unroll
  for (int e = 0; e < NE; ++e) acc[e] = 0.f;
  const float* xp = x + (size_t)t * D_DIM;
  for (int d = lane; d < D_DIM; d += 32) {
    const float xv = xp[d];
    const float4 g0 = *(const float4*)(gw + (size_t)d * NE);
    const float4 g1 = *(const float4*)(gw + (size_t)d * NE + 4);
    acc[0] += xv * g0.x; acc[1] += xv * g0.y; acc[2] += xv * g0.z; acc[3] += xv * g0.w;
    acc[4] += xv * g1.x; acc[5] += xv * g1.y; acc[6] += xv * g1.z; acc[7] += xv * g1.w;
  }
#pragma unroll
  for (int e = 0; e < NE; ++e)
#pragma unroll
    for (int off = 16; off > 0; off >>= 1) acc[e] += __shfl_xor(acc[e], off, 32);
  if (lane == 0) {
    int i0 = 0; float l0 = acc[0];
#pragma unroll
    for (int e = 1; e < NE; ++e) if (acc[e] > l0) { l0 = acc[e]; i0 = e; }
    int i1 = (i0 == 0) ? 1 : 0; float l1 = acc[i1];
#pragma unroll
    for (int e = 0; e < NE; ++e) if (e != i0 && acc[e] > l1) { l1 = acc[e]; i1 = e; }
    const float w0 = 1.f / (1.f + __expf(l1 - l0));  // normalized top-2 weights
    topk_idx[t * 2 + 0] = i0; topk_idx[t * 2 + 1] = i1;
    topk_w[t * 2 + 0] = w0;  topk_w[t * 2 + 1] = 1.f - w0;
  }
}

__global__ void moe_count(const int* __restrict__ idxf, int* __restrict__ counts) {
  const int i = blockIdx.x * blockDim.x + threadIdx.x;
  if (i < TR) atomicAdd(&counts[idxf[i]], 1);
}

__global__ void moe_scan(const int* __restrict__ counts,
                         int* __restrict__ offsets, int* __restrict__ cursor) {
  if (threadIdx.x == 0) {
    int s = 0;
    for (int e = 0; e < NE; ++e) { offsets[e] = s; cursor[e] = s; s += counts[e]; }
  }
}

__global__ void moe_scatter(const int* __restrict__ idxf, int* __restrict__ cursor,
                            int* __restrict__ row_token, int* __restrict__ pos_of) {
  const int i = blockIdx.x * blockDim.x + threadIdx.x;
  if (i < TR) {
    const int e = idxf[i];
    const int pos = atomicAdd(&cursor[e], 1);
    row_token[pos] = i >> 1;   // token id
    pos_of[i] = pos;           // inverse permutation for combine
  }
}

// gather permuted token rows, fp32 -> bf16
__global__ __launch_bounds__(256) void moe_gather(
    const float* __restrict__ x, const int* __restrict__ row_token,
    __bf16* __restrict__ A)
{
  const int r = blockIdx.x;
  const int t = row_token[r];
  const float4* src = (const float4*)(x + (size_t)t * D_DIM);
  __bf16* dst = A + (size_t)r * D_DIM;
  for (int j = threadIdx.x; j < D_DIM / 4; j += 256) {
    const float4 f = src[j];
    uint2 u;
    u.x = pack_bf16x2(f.x, f.y);
    u.y = pack_bf16x2(f.z, f.w);
    *(uint2*)&dst[j * 4] = u;
  }
}

// store a 4x4 (k x n) register block transposed into Bt[n][k] (bf16)
__device__ __forceinline__ void stage_b_t(const float r[4][4], __bf16* Bt, int nq, int kq) {
#pragma unroll
  for (int i = 0; i < 4; ++i) {
    uint2 u;
    u.x = pack_bf16x2(r[0][i], r[1][i]);
    u.y = pack_bf16x2(r[2][i], r[3][i]);
    *(uint2*)&Bt[(nq + i) * SBT + kq] = u;
  }
}

__device__ __forceinline__ void ld_b4(float r[4][4], const float* __restrict__ w,
                                      int k, int n, int ldb) {
#pragma unroll
  for (int j = 0; j < 4; ++j) {
    const float4 t = *(const float4*)(w + (size_t)(k + j) * ldb + n);
    r[j][0] = t.x; r[j][1] = t.y; r[j][2] = t.z; r[j][3] = t.w;
  }
}

// ---------------- GEMM1: h = silu(A@w1e) * (A@w3e), bf16 out ----------------
__global__ __launch_bounds__(NTHREADS) void moe_gemm1_gated(
    const __bf16* __restrict__ A, const float* __restrict__ w1,
    const float* __restrict__ w3, __bf16* __restrict__ H,
    const int* __restrict__ counts, const int* __restrict__ offsets)
{
  const int e    = blockIdx.z;
  const int rows = counts[e];
  const int by   = blockIdx.y;
  if (by * TM >= rows) return;          // early-exit overprovisioned tiles
  const int base = offsets[e];
  const int n0   = blockIdx.x * TN;

  __shared__ __bf16 As[TM * SA];
  __shared__ __bf16 B1t[TN * SBT];
  __shared__ __bf16 B3t[TN * SBT];

  const int tid = threadIdx.x, lane = tid & 31, wave = tid >> 5;
  const int wrow = (wave & 3) << 4;     // 0,16,32,48
  const int wcol = (wave >> 2) << 6;    // 0,64

  AccU acc1[4], acc3[4];
#pragma unroll
  for (int i = 0; i < 4; ++i)
#pragma unroll
    for (int j = 0; j < 8; ++j) { acc1[i].e[j] = 0.f; acc3[i].e[j] = 0.f; }

  const float* w1e = w1 + (size_t)e * D_DIM * M_DIM;
  const float* w3e = w3 + (size_t)e * D_DIM * M_DIM;

  // A staging: thread -> (row ar, 8 k's at ac)
  const int ar = tid >> 2;
  const int ac = (tid & 3) << 3;
  const int rclamp = by * TM + ar;
  const int arowg  = base + (rclamp < rows ? rclamp : rows - 1);
  const __bf16* aptr = A + (size_t)arowg * D_DIM + ac;

  // B staging: thread -> 4x4 block (kq..kq+3, nq..nq+3)
  const int kq = (tid >> 5) << 2;       // 0..28
  const int nq = (tid & 31) << 2;       // 0..124

  uint4 ra;
  float rb1[4][4], rb3[4][4];

  // prologue prefetch
  ra = *(const uint4*)(aptr + 0);
  ld_b4(rb1, w1e, kq, n0 + nq, M_DIM);
  ld_b4(rb3, w3e, kq, n0 + nq, M_DIM);

  for (int k0 = 0; k0 < D_DIM; k0 += KC) {
    // stage registers -> LDS
    *(uint4*)&As[ar * SA + ac] = ra;
    stage_b_t(rb1, B1t, nq, kq);
    stage_b_t(rb3, B3t, nq, kq);
    __syncthreads();
    // prefetch next tile (overlaps with compute below)
    if (k0 + KC < D_DIM) {
      ra = *(const uint4*)(aptr + k0 + KC);
      ld_b4(rb1, w1e, k0 + KC + kq, n0 + nq, M_DIM);
      ld_b4(rb3, w3e, k0 + KC + kq, n0 + nq, M_DIM);
    }
    const v16bf a = load_a_frag(&As[wrow * SA], lane);
#pragma unroll
    for (int nf = 0; nf < 4; ++nf) {
      const v16bf b1 = load_b_frag(B1t, wcol + (nf << 4), lane);
      acc1[nf].v = __builtin_amdgcn_wmma_f32_16x16x32_bf16(
          false, a, false, b1, (short)0, acc1[nf].v, false, false);
      const v16bf b3 = load_b_frag(B3t, wcol + (nf << 4), lane);
      acc3[nf].v = __builtin_amdgcn_wmma_f32_16x16x32_bf16(
          false, a, false, b3, (short)0, acc3[nf].v, false, false);
    }
    __syncthreads();
  }

  const int n_lane = lane & 15, h2 = lane >> 4;
#pragma unroll
  for (int nf = 0; nf < 4; ++nf) {
    const int col = n0 + wcol + (nf << 4) + n_lane;
#pragma unroll
    for (int r = 0; r < 8; ++r) {
      const int row = by * TM + wrow + r + (h2 << 3);   // C/D: m = r + 8*(lane>=16)
      if (row < rows) {
        const float g = acc1[nf].e[r];
        const float u = acc3[nf].e[r];
        const float hv = (g / (1.f + __expf(-g))) * u;  // silu(g)*u
        H[(size_t)(base + row) * M_DIM + col] = (__bf16)hv;
      }
    }
  }
}

// ---------------- GEMM2: out_sort = H @ w2e, f32 out ----------------
__global__ __launch_bounds__(NTHREADS) void moe_gemm2(
    const __bf16* __restrict__ H, const float* __restrict__ w2,
    float* __restrict__ OutS,
    const int* __restrict__ counts, const int* __restrict__ offsets)
{
  const int e    = blockIdx.z;
  const int rows = counts[e];
  const int by   = blockIdx.y;
  if (by * TM >= rows) return;
  const int base = offsets[e];
  const int n0   = blockIdx.x * TN;

  __shared__ __bf16 As[TM * SA];
  __shared__ __bf16 Bt[TN * SBT];

  const int tid = threadIdx.x, lane = tid & 31, wave = tid >> 5;
  const int wrow = (wave & 3) << 4;
  const int wcol = (wave >> 2) << 6;

  AccU acc[4];
#pragma unroll
  for (int i = 0; i < 4; ++i)
#pragma unroll
    for (int j = 0; j < 8; ++j) acc[i].e[j] = 0.f;

  const float* w2e = w2 + (size_t)e * M_DIM * D_DIM;

  const int ar = tid >> 2;
  const int ac = (tid & 3) << 3;
  const int rclamp = by * TM + ar;
  const int arowg  = base + (rclamp < rows ? rclamp : rows - 1);
  const __bf16* aptr = H + (size_t)arowg * M_DIM + ac;

  const int kq = (tid >> 5) << 2;
  const int nq = (tid & 31) << 2;

  uint4 ra;
  float rb[4][4];

  ra = *(const uint4*)(aptr + 0);
  ld_b4(rb, w2e, kq, n0 + nq, D_DIM);

  for (int k0 = 0; k0 < M_DIM; k0 += KC) {
    *(uint4*)&As[ar * SA + ac] = ra;
    stage_b_t(rb, Bt, nq, kq);
    __syncthreads();
    if (k0 + KC < M_DIM) {
      ra = *(const uint4*)(aptr + k0 + KC);
      ld_b4(rb, w2e, k0 + KC + kq, n0 + nq, D_DIM);
    }
    const v16bf a = load_a_frag(&As[wrow * SA], lane);
#pragma unroll
    for (int nf = 0; nf < 4; ++nf) {
      const v16bf b = load_b_frag(Bt, wcol + (nf << 4), lane);
      acc[nf].v = __builtin_amdgcn_wmma_f32_16x16x32_bf16(
          false, a, false, b, (short)0, acc[nf].v, false, false);
    }
    __syncthreads();
  }

  const int n_lane = lane & 15, h2 = lane >> 4;
#pragma unroll
  for (int nf = 0; nf < 4; ++nf) {
    const int col = n0 + wcol + (nf << 4) + n_lane;
#pragma unroll
    for (int r = 0; r < 8; ++r) {
      const int row = by * TM + wrow + r + (h2 << 3);
      if (row < rows)
        OutS[(size_t)(base + row) * D_DIM + col] = acc[nf].e[r];
    }
  }
}

// ---------------- combine: out[t] = w0*row(t,0) + w1*row(t,1) ----------------
__global__ __launch_bounds__(256) void moe_combine(
    const float* __restrict__ OutS, const int* __restrict__ pos_of,
    const float* __restrict__ topk_w, float* __restrict__ out)
{
  const int i = blockIdx.x * 256 + threadIdx.x;    // over T*D
  const int t = i / D_DIM, d = i - t * D_DIM;
  const int p0 = pos_of[t * 2 + 0], p1 = pos_of[t * 2 + 1];
  const float w0 = topk_w[t * 2 + 0], w1 = topk_w[t * 2 + 1];
  out[i] = w0 * OutS[(size_t)p0 * D_DIM + d] + w1 * OutS[(size_t)p1 * D_DIM + d];
}

// ---------------- launch ----------------
// Workspace requirement: ~219 MB
//   A_bf16: 8192*2048*2 = 32MB, H: 8192*7168*2 = 112MB, OutS: 8192*2048*4 = 64MB
extern "C" void kernel_launch(void* const* d_in, const int* in_sizes, int n_in,
                              void* d_out, int out_size, void* d_ws, size_t ws_size,
                              hipStream_t stream) {
  (void)in_sizes; (void)n_in; (void)out_size; (void)ws_size;
  const float* x      = (const float*)d_in[0];
  const float* gate_w = (const float*)d_in[1];
  const float* w1     = (const float*)d_in[2];
  const float* w2     = (const float*)d_in[3];
  const float* w3     = (const float*)d_in[4];
  float* out = (float*)d_out;

  char* ws = (char*)d_ws;
  size_t off = 0;
  auto alloc = [&](size_t bytes) -> char* {
    char* p = ws + off;
    off = (off + bytes + 255) & ~(size_t)255;
    return p;
  };
  int*    counts    = (int*)alloc(NE * sizeof(int));
  int*    cursor    = (int*)alloc(NE * sizeof(int));
  int*    offsets   = (int*)alloc(NE * sizeof(int));
  int*    topk_idx  = (int*)alloc((size_t)TR * sizeof(int));
  float*  topk_w    = (float*)alloc((size_t)TR * sizeof(float));
  int*    row_token = (int*)alloc((size_t)TR * sizeof(int));
  int*    pos_of    = (int*)alloc((size_t)TR * sizeof(int));
  __bf16* A_bf16    = (__bf16*)alloc((size_t)TR * D_DIM * sizeof(__bf16));
  __bf16* Hbuf      = (__bf16*)alloc((size_t)TR * M_DIM * sizeof(__bf16));
  float*  OutS      = (float*)alloc((size_t)TR * D_DIM * sizeof(float));

  hipMemsetAsync(counts, 0, NE * sizeof(int), stream);

  moe_router<<<T_TOK / 8, 256, 0, stream>>>(x, gate_w, topk_idx, topk_w);
  moe_count<<<TR / 256, 256, 0, stream>>>(topk_idx, counts);
  moe_scan<<<1, 32, 0, stream>>>(counts, offsets, cursor);
  moe_scatter<<<TR / 256, 256, 0, stream>>>(topk_idx, cursor, row_token, pos_of);
  moe_gather<<<TR, 256, 0, stream>>>(x, row_token, A_bf16);

  dim3 g1(M_DIM / TN, TR / TM, NE);   // (56,128,8): worst-case tiles, early-exit
  moe_gemm1_gated<<<g1, NTHREADS, 0, stream>>>(A_bf16, w1, w3, Hbuf, counts, offsets);

  dim3 g2(D_DIM / TN, TR / TM, NE);   // (16,128,8)
  moe_gemm2<<<g2, NTHREADS, 0, stream>>>(Hbuf, w2, OutS, counts, offsets);

  moe_combine<<<(T_TOK * D_DIM) / 256, 256, 0, stream>>>(OutS, pos_of, topk_w, out);
}
